// NeuralControlCritic_11098195493536
// MI455X (gfx1250) — matrix operand order repaired and checked by reading, not compiled, not run on hardware
//
#include <hip/hip_runtime.h>
#include <hip/hip_bf16.h>
#include <math.h>

// ---------------- problem constants (match reference) ----------------
#define BATCH      32
#define D_MODEL    1024
#define SEQ        8
#define HID        1024
#define D_INNER    2048
#define HEADDIM    64
#define NHEADS     32
#define DSTATE     128
#define DCONV      4
#define CONV_DIM   2304            // D_INNER + 2*DSTATE
#define D_IN_PROJ  4384            // 2*D_INNER + 2*DSTATE + NHEADS
#define OUT_N      1312            // REWARD*LOGITS = 32*41

typedef __attribute__((ext_vector_type(16))) _Float16 v16h;
typedef __attribute__((ext_vector_type(8)))  float    v8f;
typedef __attribute__((ext_vector_type(4)))  float    f32x4;

// ---------------- f32x4 quads -> f16 WMMA fragment ----------------
__device__ __forceinline__ v16h cvt16(f32x4 a0, f32x4 a1, f32x4 a2, f32x4 a3) {
  v16h f;
  f[0]=(_Float16)a0.x;  f[1]=(_Float16)a0.y;  f[2]=(_Float16)a0.z;  f[3]=(_Float16)a0.w;
  f[4]=(_Float16)a1.x;  f[5]=(_Float16)a1.y;  f[6]=(_Float16)a1.z;  f[7]=(_Float16)a1.w;
  f[8]=(_Float16)a2.x;  f[9]=(_Float16)a2.y;  f[10]=(_Float16)a2.z; f[11]=(_Float16)a2.w;
  f[12]=(_Float16)a3.x; f[13]=(_Float16)a3.y; f[14]=(_Float16)a3.z; f[15]=(_Float16)a3.w;
  return f;
}

// store one 16x16 C/D tile (8 strided floats per lane)
__device__ __forceinline__ void store_tile(float* __restrict__ q, const v8f& acc, int N) {
  #pragma unroll
  for (int i = 0; i < 8; ++i) q[(size_t)i * N] = acc[i];
}

// ==================================================================
// GEMM partial:  P[kc][M][N] = X(MxK) * W(NxK)^T   over one K-chunk
// grid = (ceil(N/512), M/16, kchunks), block = 256 (8 waves)
// wave w -> 16 rows x 64 cols (4 n-subtiles), A frag reused 4x.
// fp32 loads, in-register cvt to f16, v_wmma_f32_16x16x32_f16.
// ==================================================================
__global__ __launch_bounds__(256)
void gemm_partial_kernel(const float* __restrict__ X, const float* __restrict__ W,
                         float* __restrict__ P, int M, int N, int K, int kchunk) {
  const int wave  = threadIdx.x >> 5;
  const int lane  = threadIdx.x & 31;
  const int r     = lane & 15;
  const int half  = lane >> 4;
  const int m0    = blockIdx.y << 4;
  const int nbase = blockIdx.x * 512 + wave * 64;
  const int k0    = blockIdx.z * kchunk;

  // A fragment source: row m0+r, 16 floats at k+half*16
  const float* pa  = X + (size_t)(m0 + r) * K + k0 + half * 16;
  // B fragment source: column n = n0+r of output = row n of W; 16 floats at k+half*16
  const bool v0 = (nbase +  0) < N;
  const bool v1 = (nbase + 16) < N;
  const bool v2 = (nbase + 32) < N;
  const bool v3 = (nbase + 48) < N;
  const float* pb0 = W + (size_t)(nbase +  0 + r) * K + k0 + half * 16;
  const float* pb1 = W + (size_t)(nbase + 16 + r) * K + k0 + half * 16;
  const float* pb2 = W + (size_t)(nbase + 32 + r) * K + k0 + half * 16;
  const float* pb3 = W + (size_t)(nbase + 48 + r) * K + k0 + half * 16;

  v8f acc0 = {}, acc1 = {}, acc2 = {}, acc3 = {};

  for (int k = 0; k < kchunk; k += 32) {
    // ---- A fragment (16x32 f16, k = half*16 + s, same order as B) ----
    f32x4 a0 = *(const f32x4*)(pa);
    f32x4 a1 = *(const f32x4*)(pa + 4);
    f32x4 a2 = *(const f32x4*)(pa + 8);
    f32x4 a3 = *(const f32x4*)(pa + 12);
    __builtin_prefetch(pa + 256, 0, 1);
    v16h af = cvt16(a0, a1, a2, a3);

    if (v0) {
      f32x4 b0 = *(const f32x4*)(pb0), b1 = *(const f32x4*)(pb0 + 4);
      f32x4 b2 = *(const f32x4*)(pb0 + 8), b3 = *(const f32x4*)(pb0 + 12);
      __builtin_prefetch(pb0 + 256, 0, 1);
      v16h bf = cvt16(b0, b1, b2, b3);
      acc0 = __builtin_amdgcn_wmma_f32_16x16x32_f16(false, af, false, bf, (short)0, acc0, false, false);
    }
    if (v1) {
      f32x4 b0 = *(const f32x4*)(pb1), b1 = *(const f32x4*)(pb1 + 4);
      f32x4 b2 = *(const f32x4*)(pb1 + 8), b3 = *(const f32x4*)(pb1 + 12);
      __builtin_prefetch(pb1 + 256, 0, 1);
      v16h bf = cvt16(b0, b1, b2, b3);
      acc1 = __builtin_amdgcn_wmma_f32_16x16x32_f16(false, af, false, bf, (short)0, acc1, false, false);
    }
    if (v2) {
      f32x4 b0 = *(const f32x4*)(pb2), b1 = *(const f32x4*)(pb2 + 4);
      f32x4 b2 = *(const f32x4*)(pb2 + 8), b3 = *(const f32x4*)(pb2 + 12);
      __builtin_prefetch(pb2 + 256, 0, 1);
      v16h bf = cvt16(b0, b1, b2, b3);
      acc2 = __builtin_amdgcn_wmma_f32_16x16x32_f16(false, af, false, bf, (short)0, acc2, false, false);
    }
    if (v3) {
      f32x4 b0 = *(const f32x4*)(pb3), b1 = *(const f32x4*)(pb3 + 4);
      f32x4 b2 = *(const f32x4*)(pb3 + 8), b3 = *(const f32x4*)(pb3 + 12);
      __builtin_prefetch(pb3 + 256, 0, 1);
      v16h bf = cvt16(b0, b1, b2, b3);
      acc3 = __builtin_amdgcn_wmma_f32_16x16x32_f16(false, af, false, bf, (short)0, acc3, false, false);
    }
    pa += 32; pb0 += 32; pb1 += 32; pb2 += 32; pb3 += 32;
  }

  // ---- store partials: C/D layout -> element (m0+half*8+i, n0+r) = acc[i]
  float* base = P + (size_t)blockIdx.z * M * N + (size_t)(m0 + half * 8) * N + r;
  if (v0) store_tile(base + nbase +  0, acc0, N);
  if (v1) store_tile(base + nbase + 16, acc1, N);
  if (v2) store_tile(base + nbase + 32, acc2, N);
  if (v3) store_tile(base + nbase + 48, acc3, N);
}

// deterministic K-chunk reduction + bias + optional relu
__global__ __launch_bounds__(256)
void reduce_partials_kernel(const float* __restrict__ P, const float* __restrict__ bias,
                            float* __restrict__ Y, int MN, int N, int kc, int relu) {
  int i = blockIdx.x * 256 + threadIdx.x;
  if (i >= MN) return;
  float acc = 0.f;
  for (int c = 0; c < kc; ++c) acc += P[(size_t)c * MN + i];
  if (bias) acc += bias[i % N];
  if (relu) acc = fmaxf(acc, 0.f);
  Y[i] = acc;
}

// ==================================================================
// depthwise causal conv1d (k=4) + SiLU over xBC slice of zxbcdt
// ==================================================================
__global__ __launch_bounds__(256)
void conv_silu_kernel(const float* __restrict__ zxbcdt, const float* __restrict__ convw,
                      const float* __restrict__ convb, float* __restrict__ xbc) {
  int idx = blockIdx.x * 256 + threadIdx.x;          // over BATCH*CONV_DIM
  if (idx >= BATCH * CONV_DIM) return;
  int c = idx % CONV_DIM, b = idx / CONV_DIM;
  float w0 = convw[c * 4 + 0], w1 = convw[c * 4 + 1];
  float w2 = convw[c * 4 + 2], w3 = convw[c * 4 + 3];
  float bias = convb[c];
  const float* src = zxbcdt + (size_t)b * SEQ * D_IN_PROJ + D_INNER + c;
  float xv[SEQ + 3];
  xv[0] = xv[1] = xv[2] = 0.f;
  #pragma unroll
  for (int l = 0; l < SEQ; ++l) xv[3 + l] = src[(size_t)l * D_IN_PROJ];
  #pragma unroll
  for (int l = 0; l < SEQ; ++l) {
    float a = bias + xv[l] * w0 + xv[l + 1] * w1 + xv[l + 2] * w2 + xv[l + 3] * w3;
    float s = a / (1.f + expf(-a));                  // SiLU
    xbc[((size_t)b * SEQ + l) * CONV_DIM + c] = s;
  }
}

// ==================================================================
// Mamba2 selective-state scan: one block per (b,h); 64x128 state in regs
// thread t: p = t&63, n-chunk = t>>6 (32 n's). 8 sequential steps.
// ==================================================================
__global__ __launch_bounds__(256)
void ssm_scan_kernel(const float* __restrict__ zxbcdt, const float* __restrict__ xbc,
                     const float* __restrict__ dt_bias, const float* __restrict__ A_log,
                     const float* __restrict__ Dp, float* __restrict__ yout) {
  const int bh = blockIdx.x;
  const int b = bh / NHEADS, h = bh % NHEADS;
  const int t = threadIdx.x;
  const int p = t & 63, nc = t >> 6;
  __shared__ float sB[DSTATE], sC[DSTATE], red[256];

  float s[32];
  #pragma unroll
  for (int j = 0; j < 32; ++j) s[j] = 0.f;
  const float A = -expf(A_log[h]);
  const float Dh = Dp[h];
  const float dtb = dt_bias[h];

  for (int l = 0; l < SEQ; ++l) {
    const size_t row = (size_t)b * SEQ + l;
    const float* cr = xbc + row * CONV_DIM;
    if (t < DSTATE)            sB[t]          = cr[D_INNER + t];
    else if (t < 2 * DSTATE)   sC[t - DSTATE] = cr[D_INNER + t];   // C starts at 2176
    __syncthreads();

    float v = zxbcdt[row * D_IN_PROJ + (D_INNER + CONV_DIM) + h] + dtb;
    float dt = (v > 20.f) ? v : log1pf(expf(v));     // softplus
    float xt = cr[h * HEADDIM + p];
    float dA = expf(dt * A);
    float cf = dt * xt;

    const float* Bp = sB + nc * 32;
    const float* Cp = sC + nc * 32;
    float part = 0.f;
    #pragma unroll
    for (int j = 0; j < 32; ++j) {
      s[j] = s[j] * dA + cf * Bp[j];
      part += s[j] * Cp[j];
    }
    red[t] = part;
    __syncthreads();
    if (t < 64) {
      float y = red[t] + red[t + 64] + red[t + 128] + red[t + 192] + Dh * xt;
      yout[row * D_INNER + h * HEADDIM + t] = y;
    }
    __syncthreads();
  }
}

// ==================================================================
// y = y*silu(z); gated RMSNorm over 2048 dims; one block per (b,l) row
// ==================================================================
__global__ __launch_bounds__(256)
void gate_norm_kernel(const float* __restrict__ yin, const float* __restrict__ zxbcdt,
                      const float* __restrict__ normw, float* __restrict__ yout) {
  const int row = blockIdx.x;                        // 256 rows
  const int t = threadIdx.x;
  __shared__ float red[256];
  const float* y = yin + (size_t)row * D_INNER;
  const float* z = zxbcdt + (size_t)row * D_IN_PROJ; // z is first 2048
  float vals[8];
  float sq = 0.f;
  #pragma unroll
  for (int i = 0; i < 8; ++i) {
    int c = t + i * 256;
    float g = z[c];
    float yv = y[c] * (g / (1.f + expf(-g)));
    vals[i] = yv;
    sq += yv * yv;
  }
  red[t] = sq;
  __syncthreads();
  for (int sft = 128; sft > 0; sft >>= 1) {
    if (t < sft) red[t] += red[t + sft];
    __syncthreads();
  }
  float rms = rsqrtf(red[0] / (float)D_INNER + 1e-5f);
  #pragma unroll
  for (int i = 0; i < 8; ++i) {
    int c = t + i * 256;
    yout[(size_t)row * D_INNER + c] = vals[i] * rms * normw[c];
  }
}

// ---------------- host-side orchestration ----------------
struct Branch {
  const float *w1i, *b1i, *w2i, *b2i, *inproj, *convw, *convb, *dtb,
              *alog, *dd, *normw, *outproj, *w1o, *b1o, *w2o, *b2o;
};

static void run_gemm(const float* X, const float* W, const float* bias, float* Y,
                     float* part, int M, int N, int K, int kc, int relu, hipStream_t s) {
  int kchunk = K / kc;                                // all multiples of 32 by construction
  dim3 grid((N + 511) / 512, M / 16, kc);
  gemm_partial_kernel<<<grid, 256, 0, s>>>(X, W, part, M, N, K, kchunk);
  int MN = M * N;
  reduce_partials_kernel<<<(MN + 255) / 256, 256, 0, s>>>(part, bias, Y, MN, N, kc, relu);
}

static void run_branch(const float* x, const Branch& P, float* out, float* ws, hipStream_t s) {
  float* zx    = ws;                  // 256*4384      = 1,122,304 f
  float* h1    = zx    + 1122304;     // 32*8192       =   262,144 f
  float* u     = h1    + 262144;      // 32*8192
  float* xbc   = u     + 262144;      // 256*2304      =   589,824 f
  float* yssm  = xbc   + 589824;      // 256*2048      =   524,288 f
  float* ynorm = yssm  + 524288;      // 256*2048
  float* yproj = ynorm + 524288;      // 256*1024      =   262,144 f
  float* h2    = yproj + 262144;      // 32*1024       =    32,768 f
  float* part  = h2    + 32768;       // K-split partials (max ~2.25M f)

  // input MLP
  run_gemm(x,  P.w1i, P.b1i, h1, part, 32, 8192, 1024, 8, 1, s);   // relu
  run_gemm(h1, P.w2i, P.b2i, u,  part, 32, 8192, 8192, 8, 0, s);   // 268 MB stream
  // mamba2 in_proj:  (b*l,1024) @ (4384,1024)^T
  run_gemm(u,  P.inproj, nullptr, zx, part, 256, 4384, 1024, 2, 0, s);
  conv_silu_kernel<<<(BATCH * CONV_DIM) / 256, 256, 0, s>>>(zx, P.convw, P.convb, xbc);
  ssm_scan_kernel<<<BATCH * NHEADS, 256, 0, s>>>(zx, xbc, P.dtb, P.alog, P.dd, yssm);
  gate_norm_kernel<<<BATCH * SEQ, 256, 0, s>>>(yssm, zx, P.normw, ynorm);
  run_gemm(ynorm, P.outproj, nullptr, yproj, part, 256, 1024, 2048, 8, 0, s);
  // output MLP
  run_gemm(yproj, P.w1o, P.b1o, h2, part, 32, 1024, 8192, 32, 1, s); // relu
  run_gemm(h2,   P.w2o, P.b2o, out, part, 32, OUT_N, 1024, 16, 0, s);
}

extern "C" void kernel_launch(void* const* d_in, const int* in_sizes, int n_in,
                              void* d_out, int out_size, void* d_ws, size_t ws_size,
                              hipStream_t stream) {
  (void)in_sizes; (void)n_in; (void)out_size; (void)ws_size;
  const float* x = (const float*)d_in[0];
  Branch m, e;
  const float** mp = (const float**)&m;
  const float** ep = (const float**)&e;
  for (int i = 0; i < 16; ++i) {
    mp[i] = (const float*)d_in[1 + i];
    ep[i] = (const float*)d_in[17 + i];
  }
  float* out = (float*)d_out;
  float* ws = (float*)d_ws;
  run_branch(x, m, out, ws, stream);                     // main critic
  run_branch(x, e, out + BATCH * OUT_N, ws, stream);     // EMA critic (stop_gradient fwd)
}
